// GNAM_85005992722880
// MI455X (gfx1250) — 8-bit matrix operand order repaired and checked by reading, compiled, not run on hardware
//
#include <hip/hip_runtime.h>
#include <stdint.h>

#define NN 2048
#define CC 64
#define KMAX 24

typedef int v8i __attribute__((ext_vector_type(8)));


__global__ void k_clear(int4* p, size_t n_int4, unsigned* newCount) {
    size_t i = (size_t)blockIdx.x * blockDim.x + threadIdx.x;
    size_t stride = (size_t)gridDim.x * blockDim.x;
    int4 z; z.x = z.y = z.z = z.w = 0;
    for (; i < n_int4; i += stride) p[i] = z;
    if (blockIdx.x == 0 && threadIdx.x <= KMAX)
        newCount[threadIdx.x] = (threadIdx.x == 0) ? 1u : 0u;
}

__global__ void k_build(const int* __restrict__ edge, int E,
                        uint8_t* __restrict__ AT, uint8_t* __restrict__ R0) {
    int t = blockIdx.x * blockDim.x + threadIdx.x;
    if (t < E) {
        int s = edge[t];
        int d = edge[E + t];
        AT[(size_t)d * NN + s] = 1;
    }
    if (t < NN) R0[(size_t)t * NN + t] = 1;
}

__global__ void k_init(const float* __restrict__ x, const float* __restrict__ wf,
                       const float* __restrict__ bf,
                       float* __restrict__ fsums, float* __restrict__ acc) {
    int v = blockIdx.x * blockDim.x + threadIdx.x;
    if (v >= NN) return;
    float dot = 0.f;
    #pragma unroll 8
    for (int l = 0; l < CC; ++l) dot += x[v * CC + l] * wf[l];
    float bs = 0.f;
    #pragma unroll 8
    for (int l = 0; l < CC; ++l) bs += bf[l];
    fsums[v] = dot + bs;
    acc[v] = 0.f;
}


__global__ __launch_bounds__(128)
void k_bmm(const uint8_t* __restrict__ Rold, uint8_t* __restrict__ Rnew,
           const uint8_t* __restrict__ AT, const float* __restrict__ fsums,
           float* __restrict__ acc,
           const unsigned* __restrict__ prevCount, unsigned* __restrict__ curCount,
           float kdist) {
    if (*prevCount == 0u) return;

    const int lane = threadIdx.x & 31;
    const int wv   = threadIdx.x >> 5;
    const int lm   = lane & 15;
    const int half = lane >> 4;
    const int m0 = (blockIdx.y << 6) + (wv << 4);
    const int j0 = blockIdx.x << 6;

    const uint8_t* rowR = Rold + (size_t)(m0 + lm) * NN + (half << 4);
    const uint8_t* colA0 = AT + (size_t)(j0 + 0 * 16 + lm) * NN + (half << 4);
    const uint8_t* colA1 = AT + (size_t)(j0 + 1 * 16 + lm) * NN + (half << 4);
    const uint8_t* colA2 = AT + (size_t)(j0 + 2 * 16 + lm) * NN + (half << 4);
    const uint8_t* colA3 = AT + (size_t)(j0 + 3 * 16 + lm) * NN + (half << 4);

    v8i c0 = {0,0,0,0,0,0,0,0};
    v8i c1 = {0,0,0,0,0,0,0,0};
    v8i c2 = {0,0,0,0,0,0,0,0};
    v8i c3 = {0,0,0,0,0,0,0,0};

    for (int kk = 0; kk < NN; kk += 64) {
        union { v8i v; unsigned long long u[4]; } a;
        const unsigned long long* pa = (const unsigned long long*)(rowR + kk);
        a.u[0] = pa[0]; a.u[1] = pa[1]; a.u[2] = pa[4]; a.u[3] = pa[5];

        union { v8i v; int4 q[2]; } b0, b1, b2, b3;
        const int4* pb0 = (const int4*)(colA0 + kk);
        const int4* pb1 = (const int4*)(colA1 + kk);
        const int4* pb2 = (const int4*)(colA2 + kk);
        const int4* pb3 = (const int4*)(colA3 + kk);
        b0.q[0] = pb0[0]; b0.q[1] = pb0[2];
        b1.q[0] = pb1[0]; b1.q[1] = pb1[2];
        b2.q[0] = pb2[0]; b2.q[1] = pb2[2];
        b3.q[0] = pb3[0]; b3.q[1] = pb3[2];

        c0 = __builtin_amdgcn_wmma_i32_16x16x64_iu8(false, a.v, false, b0.v, c0, false, false);
        c1 = __builtin_amdgcn_wmma_i32_16x16x64_iu8(false, a.v, false, b1.v, c1, false, false);
        c2 = __builtin_amdgcn_wmma_i32_16x16x64_iu8(false, a.v, false, b2.v, c2, false, false);
        c3 = __builtin_amdgcn_wmma_i32_16x16x64_iu8(false, a.v, false, b3.v, c3, false, false);

        __builtin_prefetch(rowR + kk + 128, 0, 3);
        __builtin_prefetch(colA0 + kk + 128, 0, 3);
        __builtin_prefetch(colA2 + kk + 128, 0, 3);
    }

    unsigned cnt = 0;
    #pragma unroll
    for (int t = 0; t < 4; ++t) {
        v8i c = (t == 0) ? c0 : (t == 1) ? c1 : (t == 2) ? c2 : c3;
        #pragma unroll
        for (int r = 0; r < 8; ++r) {
            int row = m0 + r + (half << 3);
            int col = j0 + (t << 4) + lm;
            size_t idx = (size_t)row * NN + col;
            int prod = c[r];
            unsigned old = Rold[idx];
            bool isnew = (prod != 0) && (old == 0);
            Rnew[idx] = ((prod != 0) || (old != 0)) ? (uint8_t)1 : (uint8_t)0;

            float contrib = isnew ? fsums[col] : 0.0f;
            contrib += __shfl_xor(contrib, 1, 16);
            contrib += __shfl_xor(contrib, 2, 16);
            contrib += __shfl_xor(contrib, 4, 16);
            contrib += __shfl_xor(contrib, 8, 16);
            if (lm == 0) atomicAdd(&acc[row], kdist * contrib);

            unsigned long long bal = __ballot(isnew);
            if (lane == 0) cnt += (unsigned)__popcll(bal);
        }
    }
    if (lane == 0 && cnt) atomicAdd(curCount, cnt);
}


__global__ void k_reduceS(const float* __restrict__ fsums, float* __restrict__ Ssum) {
    __shared__ float sm[256];
    float s = 0.f;
    for (int i = threadIdx.x; i < NN; i += 256) s += fsums[i];
    sm[threadIdx.x] = s;
    __syncthreads();
    for (int o = 128; o > 0; o >>= 1) {
        if ((int)threadIdx.x < o) sm[threadIdx.x] += sm[threadIdx.x + o];
        __syncthreads();
    }
    if (threadIdx.x == 0) *Ssum = sm[0];
}

__global__ void k_out(const float* __restrict__ acc, const float* __restrict__ Ssum,
                      const float* __restrict__ wm, const float* __restrict__ bm,
                      float* __restrict__ out) {
    int v = blockIdx.x * blockDim.x + threadIdx.x;
    if (v < NN) out[v] = wm[0] * acc[v] + bm[0] * (*Ssum);
}


extern "C" void kernel_launch(void* const* d_in, const int* in_sizes, int n_in,
                              void* d_out, int out_size, void* d_ws, size_t ws_size,
                              hipStream_t stream) {
    (void)n_in; (void)out_size; (void)ws_size;
    const float* x  = (const float*)d_in[0];
    const float* wf = (const float*)d_in[1];
    const float* bf = (const float*)d_in[2];
    const float* wm = (const float*)d_in[3];
    const float* bm = (const float*)d_in[4];
    const int* edge = (const int*)d_in[5];
    const int E = in_sizes[5] / 2;

    char* ws = (char*)d_ws;
    uint8_t* AT = (uint8_t*)ws;
    uint8_t* R0 = AT + (size_t)NN * NN;
    uint8_t* R1 = R0 + (size_t)NN * NN;
    float* fsums = (float*)(R1 + (size_t)NN * NN);
    float* acc   = fsums + NN;
    float* Ssum  = acc + NN;
    unsigned* newCount = (unsigned*)(Ssum + 64);

    size_t n_int4 = (2 * (size_t)NN * NN) / 16;
    k_clear<<<2048, 256, 0, stream>>>((int4*)AT, n_int4, newCount);

    int maxEN = (E > NN) ? E : NN;
    k_build<<<(maxEN + 255) / 256, 256, 0, stream>>>(edge, E, AT, R0);
    k_init<<<(NN + 127) / 128, 128, 0, stream>>>(x, wf, bf, fsums, acc);

    dim3 grid(NN / 64, NN / 64);
    uint8_t* rin = R0;
    uint8_t* rout = R1;
    for (int k = 1; k <= KMAX; ++k) {
        k_bmm<<<grid, 128, 0, stream>>>(rin, rout, AT, fsums, acc,
                                        &newCount[k - 1], &newCount[k], (float)k);
        uint8_t* t = rin; rin = rout; rout = t;
    }

    k_reduceS<<<1, 256, 0, stream>>>(fsums, Ssum);
    k_out<<<(NN + 255) / 256, 256, 0, stream>>>(acc, Ssum, wm, bm, (float*)d_out);
}
